// TeacherGRU_82712480186646
// MI455X (gfx1250) — matrix operand order repaired and checked by reading, not compiled, not run on hardware
//
#include <hip/hip_runtime.h>
#include <hip/hip_bf16.h>

// ============================================================================
// Seq2seq GRU (3-layer encoder, 2-layer AR decoder + MLP head) for MI455X.
// Attention in the reference is dead code (attn_out is deleted) -> skipped.
// All GEMMs run on v_wmma_f32_16x16x32_f16 (f16 inputs, f32 accumulate).
// ============================================================================

typedef __attribute__((ext_vector_type(16))) _Float16 v16h;
typedef __attribute__((ext_vector_type(8)))  _Float16 v8h;
typedef __attribute__((ext_vector_type(8)))  float    v8f;

#define HDIM 256
#define BSZ  8192
#define TSTEPS 30
#define PHOR 60

__device__ __forceinline__ float sigmoidf_(float v) {
    return 1.0f / (1.0f + __expf(-v));
}

// A-fragment loader, contiguous k order (matches B loads):
//   lanes 0-15 (M=lane):   elems 0-15 -> K 0..15
//   lanes 16-31 (M=lane-16): elems 0-15 -> K 16..31
// i.e. two 16B chunks at k-offsets g*16 and g*16+8 (g = lane>>4).
__device__ __forceinline__ v16h load_a_frag(const _Float16* __restrict__ rowk, int g8) {
    union { v16h v; v8h h[2]; } u;
    u.h[0] = *(const v8h*)(rowk + 2 * g8);
    u.h[1] = *(const v8h*)(rowk + 2 * g8 + 8);
    return u.v;
}

// ----------------------------------------------------------------------------
// One GRU step:  h' = (1-z)*n + z*h
//   r = sigmoid(x@Wr^T + h@Ur^T + br);  z likewise
//   n = tanh(x@Wn^T + bin + r*(h@Un^T + bhn))
// Tile: workgroup = 64 rows (batch) x 32 cols (hidden), 8 waves.
// Wave (mt,nt) owns a 16x16 D tile for ALL gates -> elementwise gate math is
// local to the wave (same C layout for every accumulator).
// XDENSE: x-projection via WMMA (K=256). Else scalar FMAs over KIN<=4.
// ----------------------------------------------------------------------------
template<bool XDENSE, int KIN>
__global__ __launch_bounds__(256) void gru_step_kernel(
    const _Float16* __restrict__ Xf16,   // (B,256) f16  [XDENSE]
    const float*    __restrict__ Xs,     // small input, row stride xstride [!XDENSE]
    int xstride,
    const _Float16* __restrict__ Hprev,  // (B,256) f16
    _Float16*       __restrict__ Hnext,  // (B,256) f16
    const _Float16* __restrict__ Wih,    // (768,256) f16 [XDENSE]
    const float*    __restrict__ WihS,   // (768,KIN) f32 [!XDENSE]
    const _Float16* __restrict__ Whh,    // (768,256) f16
    const float*    __restrict__ bih,    // (768) f32
    const float*    __restrict__ bhh)    // (768) f32
{
    const int lane = threadIdx.x & 31;
    const int wid  = threadIdx.x >> 5;
    const int mt   = wid >> 1;           // 0..3 row tile
    const int nt   = wid & 1;            // 0..1 col tile
    const int r16  = lane & 15;
    const int g    = lane >> 4;
    const int g8   = g << 3;
    const int kh   = g << 4;
    const int brow = blockIdx.x * 64 + mt * 16;
    const int j0   = blockIdx.y * 32 + nt * 16;
    const int j    = j0 + r16;           // hidden column this lane owns (B/C frag)

    v8f acc_r = {}, acc_z = {}, acc_hn = {}, acc_in = {};

    // ---- recurrent part: h_{t-1} @ Whh^T for all 3 gates -------------------
    const _Float16* aH = Hprev + (brow + r16) * HDIM;
    const _Float16* wR = Whh + (      j) * HDIM + kh;
    const _Float16* wZ = Whh + (256 + j) * HDIM + kh;
    const _Float16* wN = Whh + (512 + j) * HDIM + kh;
#pragma unroll
    for (int kb = 0; kb < HDIM; kb += 32) {
        v16h a  = load_a_frag(aH + kb, g8);
        v16h br = *(const v16h*)(wR + kb);
        v16h bz = *(const v16h*)(wZ + kb);
        v16h bn = *(const v16h*)(wN + kb);
        acc_r  = __builtin_amdgcn_wmma_f32_16x16x32_f16(false, a, false, br, (short)0, acc_r,  false, false);
        acc_z  = __builtin_amdgcn_wmma_f32_16x16x32_f16(false, a, false, bz, (short)0, acc_z,  false, false);
        acc_hn = __builtin_amdgcn_wmma_f32_16x16x32_f16(false, a, false, bn, (short)0, acc_hn, false, false);
    }

    // ---- input part: x_t @ Wih^T (dense K=256) -----------------------------
    if (XDENSE) {
        const _Float16* aX = Xf16 + (brow + r16) * HDIM;
        const _Float16* uR = Wih + (      j) * HDIM + kh;
        const _Float16* uZ = Wih + (256 + j) * HDIM + kh;
        const _Float16* uN = Wih + (512 + j) * HDIM + kh;
#pragma unroll
        for (int kb = 0; kb < HDIM; kb += 32) {
            v16h a  = load_a_frag(aX + kb, g8);
            v16h br = *(const v16h*)(uR + kb);
            v16h bz = *(const v16h*)(uZ + kb);
            v16h bn = *(const v16h*)(uN + kb);
            acc_r  = __builtin_amdgcn_wmma_f32_16x16x32_f16(false, a, false, br, (short)0, acc_r,  false, false);
            acc_z  = __builtin_amdgcn_wmma_f32_16x16x32_f16(false, a, false, bz, (short)0, acc_z,  false, false);
            acc_in = __builtin_amdgcn_wmma_f32_16x16x32_f16(false, a, false, bn, (short)0, acc_in, false, false);
        }
    }

    // ---- epilogue: gate nonlinearity, per C-layout (M=i+8g, N=lane%16) -----
    const float brz = bih[j]       + bhh[j];
    const float bzz = bih[256 + j] + bhh[256 + j];
    const float bin = bih[512 + j];
    const float bhn = bhh[512 + j];

    constexpr int KK = XDENSE ? 1 : (KIN > 0 ? KIN : 1);
    float wxr[KK], wxz[KK], wxn[KK];
    if (!XDENSE) {
#pragma unroll
        for (int k = 0; k < KIN; ++k) {
            wxr[k] = WihS[(      j) * KIN + k];
            wxz[k] = WihS[(256 + j) * KIN + k];
            wxn[k] = WihS[(512 + j) * KIN + k];
        }
    }

#pragma unroll
    for (int i = 0; i < 8; ++i) {
        const int b = brow + (g << 3) + i;
        float vr = acc_r[i], vz = acc_z[i], vin = acc_in[i], vhn = acc_hn[i];
        if (!XDENSE) {
#pragma unroll
            for (int k = 0; k < KIN; ++k) {
                const float xv = Xs[(size_t)b * xstride + k];
                vr  += xv * wxr[k];
                vz  += xv * wxz[k];
                vin += xv * wxn[k];
            }
        }
        const float hp = (float)Hprev[b * HDIM + j];
        const float r  = sigmoidf_(vr + brz);
        const float z  = sigmoidf_(vz + bzz);
        const float n  = tanhf(vin + bin + r * (vhn + bhn));
        const float hn_ = (1.0f - z) * n + z * hp;
        Hnext[b * HDIM + j] = (_Float16)hn_;
    }
}

// ----------------------------------------------------------------------------
// fc1: Out = relu(X @ W^T + b), X:(B,256) f16, W:(128,256) f16 -> (B,128) f16
// ----------------------------------------------------------------------------
__global__ __launch_bounds__(256) void fc_relu_kernel(
    const _Float16* __restrict__ X,
    const _Float16* __restrict__ W,
    const float*    __restrict__ bias,
    _Float16*       __restrict__ Out)
{
    const int lane = threadIdx.x & 31;
    const int wid  = threadIdx.x >> 5;
    const int mt   = wid >> 1;
    const int nt   = wid & 1;
    const int r16  = lane & 15;
    const int g    = lane >> 4;
    const int g8   = g << 3;
    const int kh   = g << 4;
    const int brow = blockIdx.x * 64 + mt * 16;
    const int j    = blockIdx.y * 32 + nt * 16 + r16;   // 0..127

    v8f acc = {};
    const _Float16* aX = X + (brow + r16) * HDIM;
    const _Float16* wj = W + j * HDIM + kh;
#pragma unroll
    for (int kb = 0; kb < HDIM; kb += 32) {
        v16h a = load_a_frag(aX + kb, g8);
        v16h w = *(const v16h*)(wj + kb);
        acc = __builtin_amdgcn_wmma_f32_16x16x32_f16(false, a, false, w, (short)0, acc, false, false);
    }
    const float bj = bias[j];
#pragma unroll
    for (int i = 0; i < 8; ++i) {
        const int b = brow + (g << 3) + i;
        float v = acc[i] + bj;
        Out[b * 128 + j] = (_Float16)(v > 0.0f ? v : 0.0f);
    }
}

// fc2: pred[b][o] = sum_j f1[b][j]*W2[o][j] + b2[o], o in {0,1}; W2 L2-resident
__global__ __launch_bounds__(256) void fc2_kernel(
    const _Float16* __restrict__ F1,   // (B,128)
    const float*    __restrict__ W2,   // (2,128)
    const float*    __restrict__ b2,   // (2)
    float*          __restrict__ Out,  // (B,60,2)
    int step)
{
    const int b = blockIdx.x * 256 + threadIdx.x;
    float s0 = b2[0], s1 = b2[1];
    const _Float16* f = F1 + b * 128;
#pragma unroll 4
    for (int jj = 0; jj < 128; ++jj) {
        const float v = (float)f[jj];
        s0 += v * W2[jj];
        s1 += v * W2[128 + jj];
    }
    Out[(size_t)b * (PHOR * 2) + step * 2 + 0] = s0;
    Out[(size_t)b * (PHOR * 2) + step * 2 + 1] = s1;
}

__global__ void f32_to_f16_kernel(const float* __restrict__ s, _Float16* __restrict__ d, int n) {
    const int i = blockIdx.x * 256 + threadIdx.x;
    if (i < n) d[i] = (_Float16)s[i];
}

__global__ void zero_f16_kernel(_Float16* __restrict__ p, int n) {
    const int i = blockIdx.x * 256 + threadIdx.x;
    if (i < n) p[i] = (_Float16)0.0f;
}

// ============================================================================
extern "C" void kernel_launch(void* const* d_in, const int* in_sizes, int n_in,
                              void* d_out, int out_size, void* d_ws, size_t ws_size,
                              hipStream_t stream) {
    (void)in_sizes; (void)n_in; (void)out_size; (void)ws_size;

    // -------- inputs (setup_inputs dict order, nested lists flattened) ------
    const float* x     = (const float*)d_in[0];    // (8192,30,4)
    const float* Wih0  = (const float*)d_in[1];    // (768,4)
    const float* Whh0  = (const float*)d_in[2];    // (768,256)
    const float* bih0  = (const float*)d_in[3];
    const float* bhh0  = (const float*)d_in[4];
    const float* Wih1  = (const float*)d_in[5];
    const float* Whh1  = (const float*)d_in[6];
    const float* bih1  = (const float*)d_in[7];
    const float* bhh1  = (const float*)d_in[8];
    const float* Wih2  = (const float*)d_in[9];
    const float* Whh2  = (const float*)d_in[10];
    const float* bih2  = (const float*)d_in[11];
    const float* bhh2  = (const float*)d_in[12];
    // d_in[13..16] = attention params: DEAD CODE in the reference, unused.
    const float* WihD0 = (const float*)d_in[17];   // (768,2)
    const float* WhhD0 = (const float*)d_in[18];
    const float* bihD0 = (const float*)d_in[19];
    const float* bhhD0 = (const float*)d_in[20];
    const float* WihD1 = (const float*)d_in[21];   // (768,256)
    const float* WhhD1 = (const float*)d_in[22];
    const float* bihD1 = (const float*)d_in[23];
    const float* bhhD1 = (const float*)d_in[24];
    const float* fc1w  = (const float*)d_in[25];   // (128,256)
    const float* fc1b  = (const float*)d_in[26];
    const float* fc2w  = (const float*)d_in[27];   // (2,128)
    const float* fc2b  = (const float*)d_in[28];

    // -------- workspace carve-up (~48 MB total) -----------------------------
    _Float16* base = (_Float16*)d_ws;
    size_t off = 0;
    auto take = [&](size_t n) { _Float16* q = base + off; off += n; return q; };
    const size_t WSZ  = 768 * 256;
    _Float16* pWhh0  = take(WSZ);
    _Float16* pWih1  = take(WSZ);
    _Float16* pWhh1  = take(WSZ);
    _Float16* pWih2  = take(WSZ);
    _Float16* pWhh2  = take(WSZ);
    _Float16* pWhhD0 = take(WSZ);
    _Float16* pWihD1 = take(WSZ);
    _Float16* pWhhD1 = take(WSZ);
    _Float16* pFC1   = take((size_t)128 * 256);
    const size_t SLOT = (size_t)BSZ * HDIM;        // one (B,256) f16 state
    _Float16* L0 = take(2 * SLOT);                 // rolling 2-slot per layer
    _Float16* L1 = take(2 * SLOT);
    _Float16* L2 = take(2 * SLOT);
    _Float16* D0 = take(2 * SLOT);
    _Float16* D1 = take(2 * SLOT);
    _Float16* F1 = take((size_t)BSZ * 128);

    // -------- pack weights to f16 (L2-resident, reused by every step) -------
    auto cvt = [&](const float* s, _Float16* d, int n) {
        f32_to_f16_kernel<<<(n + 255) / 256, 256, 0, stream>>>(s, d, n);
    };
    cvt(Whh0,  pWhh0,  WSZ);  cvt(Wih1, pWih1, WSZ);  cvt(Whh1, pWhh1, WSZ);
    cvt(Wih2,  pWih2,  WSZ);  cvt(Whh2, pWhh2, WSZ);  cvt(WhhD0, pWhhD0, WSZ);
    cvt(WihD1, pWihD1, WSZ);  cvt(WhhD1, pWhhD1, WSZ);
    cvt(fc1w,  pFC1, 128 * 256);

    // zero initial hidden states (slot 0 of each encoder layer)
    const int zgrid = (int)((SLOT + 255) / 256);
    zero_f16_kernel<<<zgrid, 256, 0, stream>>>(L0, (int)SLOT);
    zero_f16_kernel<<<zgrid, 256, 0, stream>>>(L1, (int)SLOT);
    zero_f16_kernel<<<zgrid, 256, 0, stream>>>(L2, (int)SLOT);

    const dim3 gg(BSZ / 64, HDIM / 32), bb(256);
    auto dense = [&](const _Float16* X, const _Float16* Hp, _Float16* Hn,
                     const _Float16* Wi, const _Float16* Wh,
                     const float* bi, const float* bh) {
        gru_step_kernel<true, 0><<<gg, bb, 0, stream>>>(
            X, nullptr, 0, Hp, Hn, Wi, nullptr, Wh, bi, bh);
    };

    // -------- encoder: wavefront over layers (rolling 2-slot state) ---------
    for (int s = 0; s < TSTEPS + 2; ++s) {
        if (s < TSTEPS) {                          // layer 0, K=4 scalar x-path
            const int t = s;
            gru_step_kernel<false, 4><<<gg, bb, 0, stream>>>(
                nullptr, x + t * 4, 120,
                L0 + (t & 1) * SLOT, L0 + ((t + 1) & 1) * SLOT,
                nullptr, Wih0, pWhh0, bih0, bhh0);
        }
        if (s >= 1 && s <= TSTEPS) {               // layer 1
            const int t = s - 1;
            dense(L0 + ((t + 1) & 1) * SLOT,
                  L1 + (t & 1) * SLOT, L1 + ((t + 1) & 1) * SLOT,
                  pWih1, pWhh1, bih1, bhh1);
        }
        if (s >= 2) {                              // layer 2
            const int t = s - 2;
            dense(L1 + ((t + 1) & 1) * SLOT,
                  L2 + (t & 1) * SLOT, L2 + ((t + 1) & 1) * SLOT,
                  pWih2, pWhh2, bih2, bhh2);
        }
    }
    // final encoder states: t=29 wrote slot (30&1)=0 -> L1+0 and L2+0.

    // -------- decoder: 60 autoregressive steps ------------------------------
    float* out = (float*)d_out;
    for (int s = 0; s < PHOR; ++s) {
        // cell 0: input = last x[:, -1, :2] (step 0) or previous prediction
        const float* xin = (s == 0) ? (x + 29 * 4) : (out + (s - 1) * 2);
        const _Float16* h0p = (s == 0) ? L1 : (D0 + (s & 1) * SLOT);
        gru_step_kernel<false, 2><<<gg, bb, 0, stream>>>(
            nullptr, xin, 120,
            h0p, D0 + ((s + 1) & 1) * SLOT,
            nullptr, WihD0, pWhhD0, bihD0, bhhD0);

        // cell 1: input = new h0
        const _Float16* h1p = (s == 0) ? L2 : (D1 + (s & 1) * SLOT);
        dense(D0 + ((s + 1) & 1) * SLOT, h1p, D1 + ((s + 1) & 1) * SLOT,
              pWihD1, pWhhD1, bihD1, bhhD1);

        // head: relu(h1 @ fc1^T + b1) @ fc2^T + b2 -> out[:, s, :]
        fc_relu_kernel<<<dim3(BSZ / 64, 128 / 32), bb, 0, stream>>>(
            D1 + ((s + 1) & 1) * SLOT, pFC1, fc1b, F1);
        fc2_kernel<<<dim3(BSZ / 256), bb, 0, stream>>>(F1, fc2w, fc2b, out, s);
    }
}
